// NeuralCDE_704374636589
// MI455X (gfx1250) — compile-verified, matrix operand order checked by reading, not run on hardware
//
#include <hip/hip_runtime.h>

#define BB 512
#define TT 1024
#define CC 32
#define HH 128
#define OO 8
#define MW 128
#define MTILE 16     // batch rows per workgroup -> 32 workgroups

typedef __bf16 bf16_t;
typedef __attribute__((ext_vector_type(16))) __bf16 v16bf;
typedef __attribute__((ext_vector_type(8)))  float  v8f;

union V16U { v16bf v; uint4 q[2]; };

// native RTNE convert -> v_cvt_*bf16_f32 on gfx1250
static __device__ inline bf16_t f2bf(float f) { return (bf16_t)f; }

// A/B fragment from an LDS row: elements {off..off+7, off+16..off+23}
static __device__ inline v16bf load_frag_lds(const bf16_t* row, int off) {
  V16U t;
  t.q[0] = *(const uint4*)(row + off);
  t.q[1] = *(const uint4*)(row + off + 16);
  return t.v;
}

// Pre-packed B tile: 16 contiguous bf16 per lane
static __device__ inline v16bf load_frag_g(const bf16_t* p) {
  V16U t;
  t.q[0] = *(const uint4*)(p);
  t.q[1] = *(const uint4*)(p + 8);
  return t.v;
}

static __device__ inline v8f wmma_bf16(v16bf a, v16bf b, v8f c) {
  return __builtin_amdgcn_wmma_f32_16x16x32_bf16(false, a, false, b, (short)0,
                                                 c, false, false);
}

// Pack W[k, n] (K = 4*32 rows, N cols) into per-(nt,kt) WMMA B tiles,
// 512 bf16 per tile, in per-lane register order for v_wmma_*_16x16x32_bf16.
__global__ void ncde_pack_w(const float* __restrict__ W, bf16_t* __restrict__ pk,
                            int N, int numKT, int total) {
  int i = blockIdx.x * blockDim.x + threadIdx.x;
  if (i >= total) return;
  int e    = i & 511;
  int tile = i >> 9;
  int kt   = tile % numKT;
  int nt   = tile / numKT;
  int lane = e >> 4;
  int j    = e & 15;
  int n    = nt * 16 + (lane & 15);
  int bk   = (lane < 16) ? 0 : 8;                    // K-half per lane group
  int k    = kt * 32 + ((j < 8) ? (bk + j) : (16 + bk + (j - 8)));
  pk[i] = f2bf(W[(size_t)k * N + n]);
}

__global__ __launch_bounds__(256) void ncde_main(
    const float* __restrict__ coeffs,
    const float* __restrict__ b1,
    const float* __restrict__ b2,
    const float* __restrict__ Wi,
    const float* __restrict__ bi,
    const float* __restrict__ Wr,
    const float* __restrict__ br,
    const bf16_t* __restrict__ pkW1,   // 32 tiles  (8 nt x 4 kt)
    const bf16_t* __restrict__ pkW2,   // 1024 tiles (256 nt x 4 kt)
    float* __restrict__ out) {
  __shared__ __attribute__((aligned(16))) bf16_t zinbf[MTILE * HH]; // stage input, bf16
  __shared__ __attribute__((aligned(16))) bf16_t hbf[MTILE * MW];   // relu(zW1+b1), bf16
  __shared__ float z0s[MTILE * HH];    // fp32 state z
  __shared__ float accs[MTILE * HH];   // RK4 accumulator
  __shared__ float gs[MTILE * HH];     // stage output k_s (written exactly once/stage)
  __shared__ float dxs[MTILE * CC];
  __shared__ float b1s[MW];
  __shared__ float b2s[HH * CC];       // 4096 floats

  const int tid   = threadIdx.x;
  const int wid   = tid >> 5;
  const int ln    = tid & 31;
  const int ln15  = ln & 15;
  const int half  = ln >> 4;
  const int rbase = half * 8;          // row base for C/D layout AND K-half for A/B
  const int b0    = blockIdx.x * MTILE;

  if (tid < MW) b1s[tid] = b1[tid];
  for (int e = 0; e < 16; ++e) b2s[tid + 256 * e] = b2[tid + 256 * e];

  // z0 = coeffs[:,0,:] @ Wi + bi   (tiny [16,32]x[32,128], scalar is fine)
  for (int e = 0; e < 8; ++e) {
    int idx = tid * 8 + e;
    int m = idx >> 7, hh = idx & 127;
    float s = bi[hh];
    const float* crow = coeffs + (size_t)(b0 + m) * TT * CC;
    for (int c = 0; c < CC; ++c) s += crow[c] * Wi[c * HH + hh];
    z0s[idx]   = s;
    zinbf[idx] = f2bf(s);
  }
  __syncthreads();

  const float kc[4] = {1.f, 2.f, 2.f, 1.f};
  const float ac[3] = {0.5f, 0.5f, 1.f};

  for (int t = 0; t < TT - 1; ++t) {
    // dX for this segment
    for (int e = 0; e < 2; ++e) {
      int i = tid * 2 + e;
      int m = i >> 5, c = i & 31;
      size_t base = ((size_t)(b0 + m) * TT + t) * CC + c;
      dxs[i] = coeffs[base + CC] - coeffs[base];
    }
    __syncthreads();

    for (int s = 0; s < 4; ++s) {
      // ---- GEMM1: h = relu(zin @ W1 + b1); wave `wid` owns n-tile `wid` ----
      {
        const bf16_t* arow = zinbf + ln15 * HH;
        v16bf a0 = load_frag_lds(arow, 0 * 32 + rbase);
        v16bf a1 = load_frag_lds(arow, 1 * 32 + rbase);
        v16bf a2 = load_frag_lds(arow, 2 * 32 + rbase);
        v16bf a3 = load_frag_lds(arow, 3 * 32 + rbase);
        const bf16_t* wb = pkW1 + (size_t)wid * 4 * 512 + ln * 16;
        v8f cacc = {};
        cacc = wmma_bf16(a0, load_frag_g(wb + 0 * 512), cacc);
        cacc = wmma_bf16(a1, load_frag_g(wb + 1 * 512), cacc);
        cacc = wmma_bf16(a2, load_frag_g(wb + 2 * 512), cacc);
        cacc = wmma_bf16(a3, load_frag_g(wb + 3 * 512), cacc);
        int n = wid * 16 + ln15;
        float bias = b1s[n];
        for (int r = 0; r < 8; ++r) {
          float v = cacc[r] + bias;
          v = v > 0.f ? v : 0.f;
          hbf[(r + rbase) * MW + n] = f2bf(v);
        }
      }
      __syncthreads();

      // ---- GEMM2: u = tanh(h @ W2 + b2); contract over C into g ----
      // Tile pair (even,odd) = full c-range of one h-row -> final g, plain store.
      {
        float dxr[2][8];
        for (int r = 0; r < 8; ++r) {
          dxr[0][r] = dxs[(r + rbase) * CC + ln15];
          dxr[1][r] = dxs[(r + rbase) * CC + 16 + ln15];
        }
        const bf16_t* arow = hbf + ln15 * MW;
        v16bf af[4];
        af[0] = load_frag_lds(arow, 0 * 32 + rbase);   // hoisted: invariant over tiles
        af[1] = load_frag_lds(arow, 1 * 32 + rbase);
        af[2] = load_frag_lds(arow, 2 * 32 + rbase);
        af[3] = load_frag_lds(arow, 3 * 32 + rbase);
        for (int p = 0; p < 16; ++p) {
          int nt0 = wid * 32 + p * 2;                  // even tile; odd = nt0+1
          const bf16_t* tb = pkW2 + (size_t)nt0 * 4 * 512 + ln * 16;
          __builtin_prefetch(tb + 8 * 512, 0, 0);      // next pair -> global_prefetch_b8
          v8f c0 = {}, c1 = {};                        // two independent WMMA chains
          for (int kt = 0; kt < 4; ++kt) {
            c0 = wmma_bf16(af[kt], load_frag_g(tb + kt * 512), c0);
            c1 = wmma_bf16(af[kt], load_frag_g(tb + (4 + kt) * 512), c1);
          }
          int hh   = nt0 >> 1;
          float bs0 = b2s[nt0 * 16 + ln15];
          float bs1 = b2s[nt0 * 16 + 16 + ln15];
          for (int r = 0; r < 8; ++r) {
            float v = tanhf(c0[r] + bs0) * dxr[0][r]
                    + tanhf(c1[r] + bs1) * dxr[1][r];
            v += __shfl_xor(v, 1, 16);
            v += __shfl_xor(v, 2, 16);
            v += __shfl_xor(v, 4, 16);
            v += __shfl_xor(v, 8, 16);                 // sum all 32 c's
            if (ln15 == 0) gs[(r + rbase) * HH + hh] = v;  // final value, plain store
          }
        }
      }
      __syncthreads();

      // ---- RK4 state update ----
      for (int e = 0; e < 8; ++e) {
        int i = tid + 256 * e;
        float gv = gs[i];
        if (s == 0) accs[i] = gv; else accs[i] += kc[s] * gv;
        if (s < 3) {
          zinbf[i] = f2bf(z0s[i] + ac[s] * gv);
        } else {
          float zn = z0s[i] + accs[i] * (1.f / 6.f);
          z0s[i]   = zn;
          zinbf[i] = f2bf(zn);
        }
      }
      __syncthreads();
    }
  }

  // readout: out = zT @ Wr + br   ([16,128]x[128,8])
  if (tid < MTILE * OO) {
    int m = tid >> 3, o = tid & 7;
    float sum = br[o];
    for (int hh = 0; hh < HH; ++hh) sum += z0s[m * HH + hh] * Wr[hh * OO + o];
    out[(size_t)(b0 + m) * OO + o] = sum;
  }
}

extern "C" void kernel_launch(void* const* d_in, const int* in_sizes, int n_in,
                              void* d_out, int out_size, void* d_ws, size_t ws_size,
                              hipStream_t stream) {
  const float* coeffs = (const float*)d_in[0];
  const float* W1     = (const float*)d_in[1];
  const float* b1     = (const float*)d_in[2];
  const float* W2     = (const float*)d_in[3];
  const float* b2     = (const float*)d_in[4];
  const float* Wi     = (const float*)d_in[5];
  const float* bi     = (const float*)d_in[6];
  const float* Wr     = (const float*)d_in[7];
  const float* br     = (const float*)d_in[8];

  // workspace: [0, 32KB) packed bf16 W1 tiles; [32KB, 32KB+1MB) packed bf16 W2 tiles
  bf16_t* pkW1 = (bf16_t*)d_ws;
  bf16_t* pkW2 = (bf16_t*)((char*)d_ws + 16384 * sizeof(bf16_t));

  ncde_pack_w<<<64, 256, 0, stream>>>(W1, pkW1, 128, 4, 16384);       // 8nt x 4kt
  ncde_pack_w<<<2048, 256, 0, stream>>>(W2, pkW2, 4096, 4, 524288);   // 256nt x 4kt
  ncde_main<<<32, 256, 0, stream>>>(coeffs, b1, b2, Wi, bi, Wr, br,
                                    pkW1, pkW2, (float*)d_out);
}
